// GraphTransformer_13400297964013
// MI455X (gfx1250) — matrix operand order repaired and checked by reading, not compiled, not run on hardware
//
#include <hip/hip_runtime.h>
#include <math.h>

#define NN 50000
#define EE 800000
#define CIN 128
#define NCOLS 512          // fused [Q|K|V|S] output columns
#define MTILES (NN / 16)   // 3125, exact
#define NTILES (NCOLS / 16) // 32
#define HD 128             // heads*d, both layers

typedef __attribute__((ext_vector_type(16))) _Float16 hvec16;
typedef __attribute__((ext_vector_type(8)))  _Float16 hvec8;
typedef __attribute__((ext_vector_type(8)))  float    fvec8;

// ---------- order-preserving float<->uint encoding for atomic max ----------
__device__ __forceinline__ unsigned enc_f32(float f) {
    unsigned u = __float_as_uint(f);
    return (u & 0x80000000u) ? ~u : (u | 0x80000000u);
}
__device__ __forceinline__ float dec_f32(unsigned e) {
    return (e & 0x80000000u) ? __uint_as_float(e ^ 0x80000000u)
                             : __uint_as_float(~e);
}

// ---------- convert f32 activations to f16 row-major ----------
__global__ void cvt_f16(const float* __restrict__ in, _Float16* __restrict__ out, int n) {
    int i = blockIdx.x * blockDim.x + threadIdx.x;
    if (i < n) out[i] = (_Float16)in[i];
}

// ---------- pack 4x [128x128] f32 weights into WMMA B-operand lane layout ----------
// packed[( (ntile*4 + kt)*32 + lane )*16 + j] = W[ kbase + j + (j>>3)*8 ][ ncol ]  (f16)
// where ncol = ntile*16 + (lane&15), kbase = kt*32 + (lane>>4)*8
__global__ void pack_w(const float* __restrict__ w0, const float* __restrict__ w1,
                       const float* __restrict__ w2, const float* __restrict__ w3,
                       const float* __restrict__ b0, const float* __restrict__ b1,
                       const float* __restrict__ b2, const float* __restrict__ b3,
                       _Float16* __restrict__ pw, float* __restrict__ pb) {
    int idx = blockIdx.x * blockDim.x + threadIdx.x; // 0..4095
    if (idx >= NTILES * 4 * 32) return;
    int lane  = idx & 31;
    int kt    = (idx >> 5) & 3;
    int ntile = idx >> 7;
    int ncol  = ntile * 16 + (lane & 15);
    int kbase = kt * 32 + ((lane >> 4) << 3);
    const float* w = (ncol < 128) ? w0 : (ncol < 256) ? w1 : (ncol < 384) ? w2 : w3;
    int nc = ncol & 127;
#pragma unroll
    for (int j = 0; j < 16; ++j)
        pw[(size_t)idx * 16 + j] = (_Float16)w[(size_t)(kbase + j + ((j >> 3) << 3)) * 128 + nc];
    if (idx < NCOLS) {
        const float* b = (idx < 128) ? b0 : (idx < 256) ? b1 : (idx < 384) ? b2 : b3;
        pb[idx] = b[idx & 127];
    }
}

// ---------- WMMA GEMM: O[50000x512] = A_f16[50000x128] @ Wpacked + bias ----------
// grid (1, 3125), block 256 (8 waves). Wave w holds its A tile (4 k-fragments)
// in VGPRs and sweeps 4 n-tiles: ntile = w*4 + t.  16 static v_wmma per kernel.
__global__ __launch_bounds__(256) void wmma_gemm(const _Float16* __restrict__ A,
                                                 const _Float16* __restrict__ PW,
                                                 const float* __restrict__ PB,
                                                 float* __restrict__ O) {
    int lane  = threadIdx.x & 31;
    int wave  = threadIdx.x >> 5;
    int mtile = blockIdx.y;
    int row   = mtile * 16 + (lane & 15);
    int klo   = (lane >> 4) << 3; // 0 or 8

    // load A tile once: 4 k-step fragments (16x32 f16 each)
    hvec16 a[4];
#pragma unroll
    for (int kt = 0; kt < 4; ++kt) {
        const _Float16* ap = A + (size_t)row * CIN + kt * 32 + klo;
        hvec8 lo = *(const hvec8*)ap;
        hvec8 hi = *(const hvec8*)(ap + 16);
#pragma unroll
        for (int j = 0; j < 8; ++j) { a[kt][j] = lo[j]; a[kt][j + 8] = hi[j]; }
    }

#pragma unroll
    for (int t = 0; t < 4; ++t) {
        int ntile = wave * 4 + t;
        fvec8 acc = {};
#pragma unroll
        for (int kt = 0; kt < 4; ++kt) {
            hvec16 b = *(const hvec16*)(PW + ((size_t)(ntile * 4 + kt) * 32 + lane) * 16);
            acc = __builtin_amdgcn_wmma_f32_16x16x32_f16(
                /*neg_a=*/false, a[kt], /*neg_b=*/false, b,
                /*c_mod=*/(short)0, acc, /*reuse_a=*/false, /*reuse_b=*/false);
        }
        int col  = ntile * 16 + (lane & 15);
        float bv = PB[col];
        int r0   = mtile * 16 + ((lane >> 4) << 3);
#pragma unroll
        for (int r = 0; r < 8; ++r)
            O[(size_t)(r0 + r) * NCOLS + col] = acc[r] + bv;
    }
}

// ---------- zero attn accumulator + amax/denom ----------
__global__ void zero3(float* __restrict__ attn, unsigned* __restrict__ amax,
                      float* __restrict__ denom, int nAttn, int nNH) {
    int i = blockIdx.x * blockDim.x + threadIdx.x;
    if (i < nAttn) attn[i] = 0.f;
    if (i < nNH) { amax[i] = 0u; denom[i] = 0.f; }
}

// ---------- edge pass 1: alpha = <q[dst], k[src]> * scale ; atomic segment max ----------
__global__ void edge_alpha_max(const float* __restrict__ Q, const float* __restrict__ Kf,
                               const int* __restrict__ src, const int* __restrict__ dst,
                               float* __restrict__ alpha, unsigned* __restrict__ amax,
                               int H, int d, float scale) {
    int idx = blockIdx.x * blockDim.x + threadIdx.x;
    if (idx >= EE * H) return;
    int e = idx / H, h = idx - e * H;
    int ns = src[e], nd = dst[e];
    const float4* qp = (const float4*)(Q + (size_t)nd * NCOLS + h * d);
    const float4* kp = (const float4*)(Kf + (size_t)ns * NCOLS + h * d);
    float s = 0.f;
    for (int i = 0; i < (d >> 2); ++i) {
        float4 a = qp[i], b = kp[i];
        s += a.x * b.x + a.y * b.y + a.z * b.z + a.w * b.w;
    }
    s *= scale;
    alpha[idx] = s;
    atomicMax(amax + (size_t)nd * H + h, enc_f32(s));
}

// ---------- edge pass 2: ex = exp(alpha - amax[dst]) ; atomic segment sum ----------
__global__ void edge_exp_sum(const int* __restrict__ dst,
                             const float* __restrict__ alpha,
                             const unsigned* __restrict__ amax,
                             float* __restrict__ exb, float* __restrict__ denom, int H) {
    int idx = blockIdx.x * blockDim.x + threadIdx.x;
    if (idx >= EE * H) return;
    int e = idx / H, h = idx - e * H;
    int nd = dst[e];
    float m  = dec_f32(amax[(size_t)nd * H + h]);
    float ex = __expf(alpha[idx] - m);
    exb[idx] = ex;
    atomicAdd(denom + (size_t)nd * H + h, ex);
}

// ---------- edge pass 3: attn[dst] += (ex/denom) * v[src] ----------
// Fixed 4 chunks of 32 floats per edge (works for H=4,d=32 and H=1,d=128,
// since chunk offset c*32 == h*d + intra-head offset in both configs).
__global__ void edge_scatter(const float* __restrict__ Vf,
                             const int* __restrict__ src, const int* __restrict__ dst,
                             const float* __restrict__ exb, const float* __restrict__ denom,
                             float* __restrict__ attn, int H, int d) {
    int idx = blockIdx.x * blockDim.x + threadIdx.x;
    if (idx >= EE * 4) return;
    int e = idx >> 2, c = idx & 3;
    int h = (c * 32) / d;              // d=32 -> h=c ; d=128 -> h=0
    int ns = src[e], nd = dst[e];
    float att = exb[(size_t)e * H + h] / (denom[(size_t)nd * H + h] + 1e-16f);
    const float* vp = Vf + (size_t)ns * NCOLS + c * 32;
    float* op = attn + (size_t)nd * HD + c * 32;
#pragma unroll
    for (int i = 0; i < 32; i += 4) {
        float4 v = *(const float4*)(vp + i);
        atomicAdd(op + i + 0, att * v.x);
        atomicAdd(op + i + 1, att * v.y);
        atomicAdd(op + i + 2, att * v.z);
        atomicAdd(op + i + 3, att * v.w);
    }
}

// ---------- layer-1 finalize: hh = f16(relu(attn + S1)) ----------
__global__ void finalize1(const float* __restrict__ attn, const float* __restrict__ O,
                          _Float16* __restrict__ hh) {
    int idx = blockIdx.x * blockDim.x + threadIdx.x;
    if (idx >= NN * CIN) return;
    int n = idx >> 7, c = idx & 127;
    float v = attn[idx] + O[(size_t)n * NCOLS + 384 + c];
    hh[idx] = (_Float16)fmaxf(v, 0.f);
}

// ---------- layer-2 finalize: out = attn2 + S2 ----------
__global__ void finalize2(const float* __restrict__ attn, const float* __restrict__ O,
                          float* __restrict__ out) {
    int idx = blockIdx.x * blockDim.x + threadIdx.x;
    if (idx >= NN * CIN) return;
    int n = idx >> 7, c = idx & 127;
    out[idx] = attn[idx] + O[(size_t)n * NCOLS + 384 + c];
}

extern "C" void kernel_launch(void* const* d_in, const int* in_sizes, int n_in,
                              void* d_out, int out_size, void* d_ws, size_t ws_size,
                              hipStream_t stream) {
    const float* x  = (const float*)d_in[0];
    const int*   ei = (const int*)d_in[1];
    const int* src = ei;
    const int* dst = ei + EE;
    const float* q1w = (const float*)d_in[2],  *q1b = (const float*)d_in[3];
    const float* k1w = (const float*)d_in[4],  *k1b = (const float*)d_in[5];
    const float* v1w = (const float*)d_in[6],  *v1b = (const float*)d_in[7];
    const float* s1w = (const float*)d_in[8],  *s1b = (const float*)d_in[9];
    const float* q2w = (const float*)d_in[10], *q2b = (const float*)d_in[11];
    const float* k2w = (const float*)d_in[12], *k2b = (const float*)d_in[13];
    const float* v2w = (const float*)d_in[14], *v2b = (const float*)d_in[15];
    const float* s2w = (const float*)d_in[16], *s2b = (const float*)d_in[17];
    float* out = (float*)d_out;

    char* ws = (char*)d_ws;
    size_t off = 0;
    auto alloc = [&](size_t bytes) -> void* {
        void* p = ws + off;
        off += (bytes + 255) & ~(size_t)255;
        return p;
    };
    _Float16* xh   = (_Float16*)alloc((size_t)NN * CIN * 2);
    _Float16* hh   = (_Float16*)alloc((size_t)NN * CIN * 2);
    float*    O    = (float*)   alloc((size_t)NN * NCOLS * 4); // shared by both layers
    float*    alp  = (float*)   alloc((size_t)EE * 4 * 4);
    float*    exb  = (float*)   alloc((size_t)EE * 4 * 4);
    unsigned* amax = (unsigned*)alloc((size_t)NN * 4 * 4);
    float*    den  = (float*)   alloc((size_t)NN * 4 * 4);
    float*    attn = (float*)   alloc((size_t)NN * CIN * 4);
    _Float16* pw1  = (_Float16*)alloc((size_t)NTILES * 4 * 32 * 16 * 2);
    float*    pb1  = (float*)   alloc((size_t)NCOLS * 4);
    _Float16* pw2  = (_Float16*)alloc((size_t)NTILES * 4 * 32 * 16 * 2);
    float*    pb2  = (float*)   alloc((size_t)NCOLS * 4);

    const int T = 256;
    int nElem = NN * CIN;

    // prep
    cvt_f16<<<(nElem + T - 1) / T, T, 0, stream>>>(x, xh, nElem);
    pack_w<<<(NTILES * 4 * 32 + T - 1) / T, T, 0, stream>>>(q1w, k1w, v1w, s1w,
                                                            q1b, k1b, v1b, s1b, pw1, pb1);
    pack_w<<<(NTILES * 4 * 32 + T - 1) / T, T, 0, stream>>>(q2w, k2w, v2w, s2w,
                                                            q2b, k2b, v2b, s2b, pw2, pb2);

    // ---- layer 1 (H=4, d=32) ----
    wmma_gemm<<<dim3(1, MTILES), T, 0, stream>>>(xh, pw1, pb1, O);
    zero3<<<(nElem + T - 1) / T, T, 0, stream>>>(attn, amax, den, nElem, NN * 4);
    {
        int H = 4, d = 32, nEH = EE * H;
        float sc = 1.f / sqrtf((float)d);
        edge_alpha_max<<<(nEH + T - 1) / T, T, 0, stream>>>(O + 0, O + 128, src, dst, alp, amax, H, d, sc);
        edge_exp_sum<<<(nEH + T - 1) / T, T, 0, stream>>>(dst, alp, amax, exb, den, H);
        edge_scatter<<<(EE * 4 + T - 1) / T, T, 0, stream>>>(O + 256, src, dst, exb, den, attn, H, d);
    }
    finalize1<<<(nElem + T - 1) / T, T, 0, stream>>>(attn, O, hh);

    // ---- layer 2 (H=1, d=128) ----
    wmma_gemm<<<dim3(1, MTILES), T, 0, stream>>>(hh, pw2, pb2, O);
    zero3<<<(nElem + T - 1) / T, T, 0, stream>>>(attn, amax, den, nElem, NN * 1);
    {
        int H = 1, d = 128, nEH = EE * H;
        float sc = 1.f / sqrtf((float)d);
        edge_alpha_max<<<(nEH + T - 1) / T, T, 0, stream>>>(O + 0, O + 128, src, dst, alp, amax, H, d, sc);
        edge_exp_sum<<<(nEH + T - 1) / T, T, 0, stream>>>(dst, alp, amax, exb, den, H);
        edge_scatter<<<(EE * 4 + T - 1) / T, T, 0, stream>>>(O + 256, src, dst, exb, den, attn, H, d);
    }
    finalize2<<<(nElem + T - 1) / T, T, 0, stream>>>(attn, O, out);
}
